// Bi_Mamba_49478023250482
// MI455X (gfx1250) — matrix operand order repaired and checked by reading, not compiled, not run on hardware
//
#include <hip/hip_runtime.h>

// ---------------------------------------------------------------------------
// Bi-Mamba block for MI455X (gfx1250, wave32).
// GEMMs use v_wmma_f32_16x16x32_bf16 with LDS-staged weight tiles and a
// compile-time K so the whole K-loop unrolls (no per-wmma dscnt stalls);
// selective scan uses in-workgroup chunked parallel scan (LDS + shfl),
// software-pipelined loads. Scratch requirement: ~65 MB.
// ---------------------------------------------------------------------------

typedef __attribute__((ext_vector_type(8)))  __bf16 v8bf;
typedef __attribute__((ext_vector_type(16))) __bf16 v16bf;
typedef __attribute__((ext_vector_type(8)))  float  v8f;

#define LSEQ 16384
#define EPSN 1e-5f

__device__ __forceinline__ float silu_f(float x) { return x / (1.f + __expf(-x)); }
__device__ __forceinline__ float softplus_f(float x) {
    return (x > 20.f) ? x : log1pf(__expf(x));
}

// ---------------------------------------------------------------------------
// Generic bf16 WMMA GEMM:  C[M x N] = A[M x K] * W[N x K]^T (+bias)(+=C)
// Wave computes a 16x64 tile. grid = (M/128, N/64), block = 256 (8 waves).
// K is a template constant in {64,128,192}; N multiple of 64 (weights padded).
// The 64xK weight tile for this block is staged in LDS once (all 8 waves
// share it), row stride padded by 8 halves to spread LDS banks. The fully
// unrolled K-loop lets the scheduler hoist ds/global loads across wmmas.
// ---------------------------------------------------------------------------
template <int K>
__global__ __launch_bounds__(256) void gemm_bf16_wmma_t(
    const __bf16* __restrict__ A, int lda,
    const __bf16* __restrict__ W,
    float* __restrict__ C, int ldc,
    const float* __restrict__ bias, int resAdd)
{
    constexpr int KP = K + 8;
    __shared__ alignas(16) __bf16 sW[64 * KP];

    const int wave = threadIdx.x >> 5;
    const int lane = threadIdx.x & 31;
    const int mrow = lane & 15;   // A row within tile / B column n
    const int g    = lane >> 4;   // k-group select
    const int m0 = (blockIdx.x * 8 + wave) * 16;
    const int n0 = blockIdx.y * 64;

    // cooperative stage of the weight tile (rows n0..n0+63)
    {
        constexpr int C8 = K >> 3;           // 8-half chunks per row
        constexpr int NCH = 64 * C8;
#pragma unroll
        for (int base = 0; base < NCH; base += 256) {
            const int idx = base + threadIdx.x;
            if (base + 256 <= NCH || idx < NCH) {
                const int r = idx / C8;
                const int c = (idx - r * C8) * 8;
                *(v8bf*)(&sW[r * KP + c]) =
                    *(const v8bf*)(W + (size_t)(n0 + r) * K + c);
            }
        }
    }
    __syncthreads();

    v8f acc[4];
#pragma unroll
    for (int j = 0; j < 4; ++j)
#pragma unroll
        for (int r = 0; r < 8; ++r) acc[j][r] = 0.f;

    const __bf16* arow = A + (size_t)(m0 + mrow) * lda;
#pragma unroll
    for (int k0 = 0; k0 < K; k0 += 32) {
        // issue all loads (A from global, 4 B fragments from LDS) up front
        v8bf a0 = *(const v8bf*)(arow + k0 + g * 8);
        v8bf a1 = *(const v8bf*)(arow + k0 + 16 + g * 8);
        v8bf b0[4], b1[4];
#pragma unroll
        for (int j = 0; j < 4; ++j) {
            const __bf16* br = &sW[(j * 16 + mrow) * KP + k0 + g * 8];
            b0[j] = *(const v8bf*)br;
            b1[j] = *(const v8bf*)(br + 16);
        }
        v16bf a = __builtin_shufflevector(a0, a1,
                    0,1,2,3,4,5,6,7,8,9,10,11,12,13,14,15);
#pragma unroll
        for (int j = 0; j < 4; ++j) {
            v16bf b = __builtin_shufflevector(b0[j], b1[j],
                        0,1,2,3,4,5,6,7,8,9,10,11,12,13,14,15);
            acc[j] = __builtin_amdgcn_wmma_f32_16x16x32_bf16(
                         false, a, false, b, (short)0, acc[j], false, false);
        }
    }
    const int rowoff = g * 8;
#pragma unroll
    for (int j = 0; j < 4; ++j) {
        const int n = n0 + j * 16 + mrow;
        const float bv = bias ? bias[n] : 0.f;
#pragma unroll
        for (int r = 0; r < 8; ++r) {
            float* cp = C + (size_t)(m0 + rowoff + r) * ldc + n;
            float v = acc[j][r] + bv;
            if (resAdd) v += *cp;
            *cp = v;
        }
    }
}

// ---------------------------------------------------------------------------
// 3x3 conv (64->64) as implicit GEMM over 9 taps, pixel-major activations.
// Wave: 16 consecutive pixels in an image row x 64 out channels.
// Out[p][o] = sum + bias[o] + resid[p][o].  grid = 128 blocks, block = 256.
// ---------------------------------------------------------------------------
__global__ __launch_bounds__(256) void conv3x3_wmma(
    const __bf16* __restrict__ A,    // L x 64 bf16 (pixel-major)
    const __bf16* __restrict__ Wr,   // 9 x 64 x 64 bf16 (tap, o, i)
    const float* __restrict__ bias,
    const float* __restrict__ resid, // L x 64 f32
    float* __restrict__ Out)         // L x 64 f32
{
    const int wave = threadIdx.x >> 5;
    const int lane = threadIdx.x & 31;
    const int mrow = lane & 15;
    const int g    = lane >> 4;
    const int t  = blockIdx.x * 8 + wave;  // 1024 pixel tiles
    const int y0 = t >> 3;
    const int x0 = (t & 7) * 16;

    v8f acc[4];
#pragma unroll
    for (int j = 0; j < 4; ++j)
#pragma unroll
        for (int r = 0; r < 8; ++r) acc[j][r] = 0.f;

#pragma unroll
    for (int tap = 0; tap < 9; ++tap) {
        const int dy = tap / 3 - 1, dx = tap % 3 - 1;
        const int yy = y0 + dy;
        const int xx = x0 + mrow + dx;
        const bool ok = (yy >= 0) && (yy < 128) && (xx >= 0) && (xx < 128);
        const __bf16* arow = A + (size_t)((yy << 7) + xx) * 64;
#pragma unroll
        for (int kk = 0; kk < 2; ++kk) {
            const int k0 = kk * 32;
            v8bf a0, a1;
            if (ok) {
                a0 = *(const v8bf*)(arow + k0 + g * 8);
                a1 = *(const v8bf*)(arow + k0 + 16 + g * 8);
            } else {
#pragma unroll
                for (int i = 0; i < 8; ++i) { a0[i] = (__bf16)0.f; a1[i] = (__bf16)0.f; }
            }
            // issue all 8 B-fragment loads before the wmma chain
            v8bf b0[4], b1[4];
#pragma unroll
            for (int j = 0; j < 4; ++j) {
                const __bf16* wrow = Wr + ((size_t)tap * 64 + j * 16 + mrow) * 64 + k0 + g * 8;
                b0[j] = *(const v8bf*)wrow;
                b1[j] = *(const v8bf*)(wrow + 16);
            }
            v16bf a = __builtin_shufflevector(a0, a1,
                        0,1,2,3,4,5,6,7,8,9,10,11,12,13,14,15);
#pragma unroll
            for (int j = 0; j < 4; ++j) {
                v16bf b = __builtin_shufflevector(b0[j], b1[j],
                            0,1,2,3,4,5,6,7,8,9,10,11,12,13,14,15);
                acc[j] = __builtin_amdgcn_wmma_f32_16x16x32_bf16(
                             false, a, false, b, (short)0, acc[j], false, false);
            }
        }
    }
    const int rowoff = g * 8;
#pragma unroll
    for (int j = 0; j < 4; ++j) {
        const int n = j * 16 + mrow;
        const float bv = bias[n];
#pragma unroll
        for (int r = 0; r < 8; ++r) {
            const int p = (y0 << 7) + x0 + rowoff + r;
            Out[(size_t)p * 64 + n] = acc[j][r] + bv + resid[(size_t)p * 64 + n];
        }
    }
}

// ---------------------------------------------------------------------------
// Selective scan, chunked within one workgroup.
// blockIdx.x = channel d (128). 256 threads = 16 half-waves = 16 chunks of
// 1024 steps; lane-within-half = state n (16). Loads for step l+1 are issued
// while step l's exp/fma chain executes (branch-free clamped prefetch).
// ---------------------------------------------------------------------------
__global__ __launch_bounds__(256) void scan_kernel(
    const float*  __restrict__ xd,    // L x 192 : [0:128]=dt_raw, [128:144]=B, [144:160]=C
    const __bf16* __restrict__ xc,    // L x 128 : u (post conv+silu)
    const float*  __restrict__ xz,    // L x 256 : z at col 128+d
    const float*  __restrict__ A_log, // 128 x 16
    const float*  __restrict__ Dp,    // 128
    const float*  __restrict__ dtb,   // 128
    float* __restrict__ y,            // L x 128
    int flip)
{
    const int d     = blockIdx.x;
    const int tid   = threadIdx.x;
    const int chunk = tid >> 4;
    const int n     = tid & 15;
    const int CH    = LSEQ / 16;
    const int l0    = chunk * CH;

    const float An  = -__expf(A_log[d * 16 + n]);
    const float bdt = dtb[d];
    const float Dd  = Dp[d];

    // pass 1: per-chunk cumulative dA product and end state (h_in = 0)
    float p = 1.f, hend = 0.f;
    {
        float nx_dt = xd[(size_t)l0 * 192 + d];
        float nx_B  = xd[(size_t)l0 * 192 + 128 + n];
        float nx_u  = (float)xc[(size_t)l0 * 128 + d];
        for (int l = l0; l < l0 + CH; ++l) {
            const float c_dt = nx_dt, c_B = nx_B, c_u = nx_u;
            const int ln = (l + 1 < LSEQ) ? (l + 1) : l;   // clamped prefetch
            nx_dt = xd[(size_t)ln * 192 + d];
            nx_B  = xd[(size_t)ln * 192 + 128 + n];
            nx_u  = (float)xc[(size_t)ln * 128 + d];
            const float dt = softplus_f(c_dt + bdt);
            const float dA = __expf(dt * An);
            hend = dA * hend + dt * c_B * c_u;
            p *= dA;
        }
    }
    __shared__ float sP[16][16], sH[16][16], sIn[16][16];
    sP[chunk][n] = p; sH[chunk][n] = hend;
    __syncthreads();
    if (tid < 16) {            // lane = state n, sequential over 16 chunks
        float h = 0.f;
        for (int c = 0; c < 16; ++c) {
            sIn[c][tid] = h;
            h = sP[c][tid] * h + sH[c][tid];
        }
    }
    __syncthreads();

    // pass 2: replay with correct initial state, emit gated output
    float h = sIn[chunk][n];
    {
        float nx_dt = xd[(size_t)l0 * 192 + d];
        float nx_B  = xd[(size_t)l0 * 192 + 128 + n];
        float nx_C  = xd[(size_t)l0 * 192 + 144 + n];
        float nx_u  = (float)xc[(size_t)l0 * 128 + d];
        for (int l = l0; l < l0 + CH; ++l) {
            const float c_dt = nx_dt, c_B = nx_B, c_C = nx_C, c_u = nx_u;
            const int ln = (l + 1 < LSEQ) ? (l + 1) : l;
            nx_dt = xd[(size_t)ln * 192 + d];
            nx_B  = xd[(size_t)ln * 192 + 128 + n];
            nx_C  = xd[(size_t)ln * 192 + 144 + n];
            nx_u  = (float)xc[(size_t)ln * 128 + d];
            const float dt = softplus_f(c_dt + bdt);
            const float dA = __expf(dt * An);
            h = dA * h + dt * c_B * c_u;
            float v = h * c_C;
            v += __shfl_xor(v, 1, 32);
            v += __shfl_xor(v, 2, 32);
            v += __shfl_xor(v, 4, 32);
            v += __shfl_xor(v, 8, 32);
            if (n == 0) {
                const int lo = flip ? (LSEQ - 1 - l) : l;
                const float zz = xz[(size_t)lo * 256 + 128 + d];
                y[(size_t)lo * 128 + d] = (v + Dd * c_u) * silu_f(zz);
            }
        }
    }
}

// ---------------------------------------------------------------------------
// Elementwise / norm / prep kernels
// ---------------------------------------------------------------------------

// copy f32 (Rsrc x Csrc) into bf16 (Rdst x Cdst), zero-padded
__global__ void prep_pad(const float* __restrict__ src, __bf16* __restrict__ dst,
                         int Rsrc, int Csrc, int Cdst, int Rdst)
{
    int idx = blockIdx.x * 256 + threadIdx.x;
    int total = Rdst * Cdst;
    if (idx >= total) return;
    int r = idx / Cdst, c = idx % Cdst;
    float v = (r < Rsrc && c < Csrc) ? src[(size_t)r * Csrc + c] : 0.f;
    dst[idx] = (__bf16)v;
}

// fused (dt,B,C) projection weights: 4 units x 192 rows x 128 cols
__global__ void prep_comb(const float* __restrict__ dtw,   // (4,128,4)
                          const float* __restrict__ xproj, // (4,36,128)
                          __bf16* __restrict__ dst)        // (4,192,128)
{
    int idx = blockIdx.x * 256 + threadIdx.x;
    if (idx >= 4 * 192 * 128) return;
    int u = idx / (192 * 128);
    int row = (idx / 128) % 192;
    int k = idx % 128;
    const float* xp = xproj + (size_t)u * 36 * 128;
    float v;
    if (row < 128) {
        const float* dw = dtw + ((size_t)u * 128 + row) * 4;
        v = dw[0] * xp[0 * 128 + k] + dw[1] * xp[1 * 128 + k]
          + dw[2] * xp[2 * 128 + k] + dw[3] * xp[3 * 128 + k];
    } else if (row < 144) v = xp[(size_t)(4 + row - 128) * 128 + k];
    else if (row < 160)   v = xp[(size_t)(20 + row - 144) * 128 + k];
    else                  v = 0.f;
    dst[idx] = (__bf16)v;
}

// resconv OIHW -> (tap, o, i) bf16
__global__ void prep_res(const float* __restrict__ src, __bf16* __restrict__ dst)
{
    int idx = blockIdx.x * 256 + threadIdx.x;
    if (idx >= 9 * 64 * 64) return;
    int tap = idx / 4096, o = (idx / 64) % 64, i = idx % 64;
    dst[idx] = (__bf16)src[((size_t)o * 64 + i) * 9 + tap];
}

// input LN (norm1) -> xnorm bf16; then LN (pe_norm) -> h, res (f32)
__global__ __launch_bounds__(256) void ln_input(
    const float* __restrict__ in, const float* __restrict__ w1, const float* __restrict__ b1,
    const float* __restrict__ wp, const float* __restrict__ bp,
    __bf16* __restrict__ xnorm, float* __restrict__ h, float* __restrict__ res)
{
    int p = blockIdx.x * 256 + threadIdx.x;
    float v[64];
    float mu = 0.f;
    for (int c = 0; c < 64; ++c) { v[c] = in[((size_t)c << 14) + p]; mu += v[c]; }
    mu *= (1.f / 64.f);
    float var = 0.f;
    for (int c = 0; c < 64; ++c) { float d = v[c] - mu; var += d * d; }
    float rs = rsqrtf(var * (1.f / 64.f) + EPSN);
    for (int c = 0; c < 64; ++c) {
        v[c] = (v[c] - mu) * rs * w1[c] + b1[c];
        xnorm[(size_t)p * 64 + c] = (__bf16)v[c];
    }
    mu = 0.f;
    for (int c = 0; c < 64; ++c) mu += v[c];
    mu *= (1.f / 64.f);
    var = 0.f;
    for (int c = 0; c < 64; ++c) { float d = v[c] - mu; var += d * d; }
    rs = rsqrtf(var * (1.f / 64.f) + EPSN);
    for (int c = 0; c < 64; ++c) {
        float s = (v[c] - mu) * rs * wp[c] + bp[c];
        h[(size_t)p * 64 + c] = s;
        res[(size_t)p * 64 + c] = s;
    }
}

__global__ __launch_bounds__(256) void rmsnorm_k(
    const float* __restrict__ x, const float* __restrict__ w, __bf16* __restrict__ dst)
{
    int p = blockIdx.x * 256 + threadIdx.x;
    const float* row = x + (size_t)p * 64;
    float s = 0.f;
    for (int c = 0; c < 64; ++c) { float v = row[c]; s += v * v; }
    float rs = rsqrtf(s * (1.f / 64.f) + EPSN);
    for (int c = 0; c < 64; ++c) dst[(size_t)p * 64 + c] = (__bf16)(row[c] * rs * w[c]);
}

// causal conv1d (K=4) + silu for forward and (flipped) backward direction
__global__ __launch_bounds__(256) void conv1d_silu(
    const float* __restrict__ xz,  // L x 256, x in cols 0..127
    const float* __restrict__ cw,  // (2,128,4)
    const float* __restrict__ cb,  // (2,128)
    __bf16* __restrict__ xcf, __bf16* __restrict__ xcb)
{
    int idx = blockIdx.x * 256 + threadIdx.x;
    int l = idx >> 7, d = idx & 127;
    const float* wf = cw + (size_t)d * 4;
    const float* wb = cw + 512 + (size_t)d * 4;
    float sf = cb[d], sb = cb[128 + d];
#pragma unroll
    for (int j = 0; j < 4; ++j) {
        int i = l - 3 + j;
        if (i >= 0) {
            sf += wf[j] * xz[(size_t)i * 256 + d];
            sb += wb[j] * xz[(size_t)(LSEQ - 1 - i) * 256 + d];
        }
    }
    xcf[idx] = (__bf16)silu_f(sf);
    xcb[idx] = (__bf16)silu_f(sb);
}

__global__ void add_bf(const float* __restrict__ a, const float* __restrict__ b,
                       __bf16* __restrict__ dst)
{
    int idx = blockIdx.x * 256 + threadIdx.x;
    dst[idx] = (__bf16)(a[idx] + b[idx]);
}

// depthwise 3x3 over 64 channels (pixel-major, stride 64) + bias
__global__ __launch_bounds__(256) void dwconv64(
    const float* __restrict__ in, const float* __restrict__ w,  // (64,9)
    const float* __restrict__ b, float* __restrict__ out)
{
    int idx = blockIdx.x * 256 + threadIdx.x;
    int p = idx >> 6, c = idx & 63;
    int py = p >> 7, px = p & 127;
    float s = b[c];
#pragma unroll
    for (int t = 0; t < 9; ++t) {
        int yy = py + t / 3 - 1, xx = px + t % 3 - 1;
        if (yy >= 0 && yy < 128 && xx >= 0 && xx < 128)
            s += w[c * 9 + t] * in[(size_t)((yy << 7) + xx) * 64 + c];
    }
    out[(size_t)p * 64 + c] = s;
}

// x = input + dwres * x_mamba ; then LN (norm2) -> xn bf16
__global__ __launch_bounds__(256) void compose_ln2(
    const float* __restrict__ in, const float* __restrict__ dwres,
    const float* __restrict__ xmamba, const float* __restrict__ w2,
    const float* __restrict__ b2, float* __restrict__ xpm, __bf16* __restrict__ xn)
{
    int p = blockIdx.x * 256 + threadIdx.x;
    float v[64];
    float mu = 0.f;
    for (int c = 0; c < 64; ++c) {
        float x = in[((size_t)c << 14) + p]
                + dwres[(size_t)p * 64 + c] * xmamba[(size_t)p * 64 + c];
        v[c] = x;
        xpm[(size_t)p * 64 + c] = x;
        mu += x;
    }
    mu *= (1.f / 64.f);
    float var = 0.f;
    for (int c = 0; c < 64; ++c) { float d = v[c] - mu; var += d * d; }
    float rs = rsqrtf(var * (1.f / 64.f) + EPSN);
    for (int c = 0; c < 64; ++c)
        xn[(size_t)p * 64 + c] = (__bf16)((v[c] - mu) * rs * w2[c] + b2[c]);
}

// depthwise 3x3 on g (stride 384, 340 channels), gelu(g1)*g2 -> bf16 (stride 192)
__global__ __launch_bounds__(256) void ffn_gate(
    const float* __restrict__ g, const float* __restrict__ w,  // (340,9)
    __bf16* __restrict__ gg)
{
    int idx = blockIdx.x * 256 + threadIdx.x;
    int p = idx / 192, c = idx % 192;
    if (c >= 170) { gg[idx] = (__bf16)0.f; return; }
    int py = p >> 7, px = p & 127;
    float s1 = 0.f, s2 = 0.f;
#pragma unroll
    for (int t = 0; t < 9; ++t) {
        int yy = py + t / 3 - 1, xx = px + t % 3 - 1;
        if (yy >= 0 && yy < 128 && xx >= 0 && xx < 128) {
            const float* base = g + (size_t)((yy << 7) + xx) * 384;
            s1 += w[c * 9 + t] * base[c];
            s2 += w[(170 + c) * 9 + t] * base[170 + c];
        }
    }
    float ge = 0.5f * s1 * (1.f + erff(s1 * 0.70710678f));
    gg[idx] = (__bf16)(ge * s2);
}

// out[c][p] = xpm[p][c] + ffo[p][c]   (pixel-major -> NCHW)
__global__ void final_out(const float* __restrict__ xpm, const float* __restrict__ ffo,
                          float* __restrict__ out)
{
    int idx = blockIdx.x * 256 + threadIdx.x;
    int c = idx >> 14, p = idx & (LSEQ - 1);
    out[((size_t)c << 14) + p] = xpm[(size_t)p * 64 + c] + ffo[(size_t)p * 64 + c];
}

// ---------------------------------------------------------------------------
// Workspace layout (bytes). Peak ~65 MB.
// ---------------------------------------------------------------------------
#define MB ((size_t)1 << 20)
#define OFF_WIN    ((size_t)0)                    // 2*256*64 bf16
#define OFF_WCOMB  ((size_t)0x10000)              // 4*192*128 bf16
#define OFF_WOUT   (OFF_WCOMB + 196608)
#define OFF_WRES   (OFF_WOUT + 32768)
#define OFF_WDW1   (OFF_WRES + 73728)
#define OFF_WFI    (OFF_WDW1 + 8192)
#define OFF_WFO    (OFF_WFI + 49152)
#define OFF_XNORM  (1 * MB)    // bf16 L*64
#define OFF_H      (3 * MB)    // f32  L*64
#define OFF_RES    (7 * MB)    // f32  L*64
#define OFF_RMS    (11 * MB)   // bf16 L*64
#define OFF_XZ     (13 * MB)   // f32  L*256
#define OFF_YSUM   (13 * MB)   // bf16 L*128 (reuses XZ after scans)
#define OFF_T1     (13 * MB)   // late: f32 L*64
#define OFF_DWRES  (17 * MB)   // late: f32 L*64
#define OFF_XPM    (21 * MB)   // late: f32 L*64
#define OFF_XNBF   (25 * MB)   // late: bf16 L*64
#define OFF_XCF    (29 * MB)   // bf16 L*128
#define OFF_XCB    (33 * MB)   // bf16 L*128
#define OFF_XD     (37 * MB)   // f32  L*192
#define OFF_YF     (49 * MB)   // f32  L*128
#define OFF_YB     (57 * MB)   // f32  L*128
#define OFF_G      (29 * MB)   // late: f32 L*384 (reuses XCF/XCB/XD/YF)
#define OFF_GG     (53 * MB)   // late: bf16 L*192
#define OFF_FFO    (59 * MB)   // late: f32 L*64

static inline void launch_gemm(int K, dim3 grid,
                               const __bf16* A, int lda, const __bf16* W,
                               float* C, int ldc, const float* bias, int resAdd,
                               hipStream_t stream)
{
    switch (K) {
    case 64:
        gemm_bf16_wmma_t<64><<<grid, 256, 0, stream>>>(A, lda, W, C, ldc, bias, resAdd);
        break;
    case 128:
        gemm_bf16_wmma_t<128><<<grid, 256, 0, stream>>>(A, lda, W, C, ldc, bias, resAdd);
        break;
    default:
        gemm_bf16_wmma_t<192><<<grid, 256, 0, stream>>>(A, lda, W, C, ldc, bias, resAdd);
        break;
    }
}

extern "C" void kernel_launch(void* const* d_in, const int* in_sizes, int n_in,
                              void* d_out, int out_size, void* d_ws, size_t ws_size,
                              hipStream_t stream)
{
    (void)in_sizes; (void)n_in; (void)out_size; (void)ws_size;
    const float* input = (const float*)d_in[0];
    const float* n1w = (const float*)d_in[1];
    const float* n1b = (const float*)d_in[2];
    const float* n2w = (const float*)d_in[3];
    const float* n2b = (const float*)d_in[4];
    const float* pew = (const float*)d_in[5];
    const float* peb = (const float*)d_in[6];
    const float* mnw = (const float*)d_in[7];
    const float* inprojw = (const float*)d_in[8];
    const float* convw = (const float*)d_in[9];
    const float* convb = (const float*)d_in[10];
    const float* xprojw = (const float*)d_in[11];
    const float* dtw = (const float*)d_in[12];
    const float* dtbp = (const float*)d_in[13];
    const float* alog = (const float*)d_in[14];
    const float* Dpp = (const float*)d_in[15];
    const float* outprojw = (const float*)d_in[16];
    const float* normfw = (const float*)d_in[17];
    const float* resw = (const float*)d_in[18];
    const float* resb = (const float*)d_in[19];
    const float* dw1w = (const float*)d_in[20];
    const float* dw1b = (const float*)d_in[21];
    const float* dw2w = (const float*)d_in[22];
    const float* dw2b = (const float*)d_in[23];
    const float* fiw = (const float*)d_in[24];
    const float* fdw = (const float*)d_in[25];
    const float* fow = (const float*)d_in[26];

    char* ws = (char*)d_ws;
    __bf16* w_in   = (__bf16*)(ws + OFF_WIN);
    __bf16* w_comb = (__bf16*)(ws + OFF_WCOMB);
    __bf16* w_out  = (__bf16*)(ws + OFF_WOUT);
    __bf16* w_res  = (__bf16*)(ws + OFF_WRES);
    __bf16* w_dw1  = (__bf16*)(ws + OFF_WDW1);
    __bf16* w_fi   = (__bf16*)(ws + OFF_WFI);
    __bf16* w_fo   = (__bf16*)(ws + OFF_WFO);
    __bf16* xnorm  = (__bf16*)(ws + OFF_XNORM);
    float*  h      = (float*)(ws + OFF_H);
    float*  res    = (float*)(ws + OFF_RES);
    __bf16* rms    = (__bf16*)(ws + OFF_RMS);
    float*  xz     = (float*)(ws + OFF_XZ);
    __bf16* ysum   = (__bf16*)(ws + OFF_YSUM);
    float*  t1     = (float*)(ws + OFF_T1);
    float*  dwres  = (float*)(ws + OFF_DWRES);
    float*  xpm    = (float*)(ws + OFF_XPM);
    __bf16* xn     = (__bf16*)(ws + OFF_XNBF);
    __bf16* xcf    = (__bf16*)(ws + OFF_XCF);
    __bf16* xcb    = (__bf16*)(ws + OFF_XCB);
    float*  xd     = (float*)(ws + OFF_XD);
    float*  yf     = (float*)(ws + OFF_YF);
    float*  yb     = (float*)(ws + OFF_YB);
    float*  g      = (float*)(ws + OFF_G);
    __bf16* gg     = (__bf16*)(ws + OFF_GG);
    float*  ffo    = (float*)(ws + OFF_FFO);

    // -------- weight prep (small) --------
    prep_pad<<<(512 * 64 + 255) / 256, 256, 0, stream>>>(inprojw, w_in, 512, 64, 64, 512);
    prep_comb<<<(4 * 192 * 128 + 255) / 256, 256, 0, stream>>>(dtw, xprojw, w_comb);
    prep_pad<<<(128 * 128 + 255) / 256, 256, 0, stream>>>(outprojw, w_out, 128, 128, 128, 128);
    prep_res<<<(9 * 4096 + 255) / 256, 256, 0, stream>>>(resw, w_res);
    prep_pad<<<(64 * 64 + 255) / 256, 256, 0, stream>>>(dw1w, w_dw1, 64, 64, 64, 64);
    prep_pad<<<(384 * 64 + 255) / 256, 256, 0, stream>>>(fiw, w_fi, 340, 64, 64, 384);
    prep_pad<<<(64 * 192 + 255) / 256, 256, 0, stream>>>(fow, w_fo, 64, 170, 192, 64);

    // -------- input norms --------
    ln_input<<<64, 256, 0, stream>>>(input, n1w, n1b, pew, peb, xnorm, h, res);

    // -------- bimamba blocks --------
    for (int dep = 0; dep < 2; ++dep) {
        rmsnorm_k<<<64, 256, 0, stream>>>(h, mnw + dep * 64, rms);
        launch_gemm(64, dim3(128, 4), rms, 64, w_in + (size_t)dep * 256 * 64,
                    xz, 256, nullptr, 0, stream);
        conv1d_silu<<<8192, 256, 0, stream>>>(
            xz, convw + (size_t)dep * 1024, convb + (size_t)dep * 256, xcf, xcb);
        for (int dir = 0; dir < 2; ++dir) {
            const __bf16* xc = dir ? xcb : xcf;
            const int u = dep * 2 + dir;
            launch_gemm(128, dim3(128, 3), xc, 128, w_comb + (size_t)u * 192 * 128,
                        xd, 192, nullptr, 0, stream);
            scan_kernel<<<128, 256, 0, stream>>>(
                xd, xc, xz, alog + (size_t)u * 2048, Dpp + (size_t)u * 128,
                dtbp + (size_t)u * 128, dir ? yb : yf, dir);
        }
        add_bf<<<8192, 256, 0, stream>>>(yf, yb, ysum);
        launch_gemm(128, dim3(128, 1), ysum, 128, w_out + (size_t)dep * 64 * 128,
                    h, 64, nullptr, 1, stream);
    }

    // -------- resconv + residual -> x_mamba (into h) --------
    rmsnorm_k<<<64, 256, 0, stream>>>(h, normfw, rms);
    conv3x3_wmma<<<128, 256, 0, stream>>>(rms, w_res, resb, res, h);

    // -------- dw path --------
    launch_gemm(64, dim3(128, 1), xnorm, 64, w_dw1, t1, 64, dw1b, 0, stream);
    dwconv64<<<4096, 256, 0, stream>>>(t1, dw2w, dw2b, dwres);
    compose_ln2<<<64, 256, 0, stream>>>(input, dwres, h, n2w, n2b, xpm, xn);

    // -------- FFN --------
    launch_gemm(64, dim3(128, 6), xn, 64, w_fi, g, 384, nullptr, 0, stream);
    ffn_gate<<<12288, 256, 0, stream>>>(g, fdw, gg);
    launch_gemm(192, dim3(128, 1), gg, 192, w_fo, ffo, 64, nullptr, 0, stream);

    final_out<<<4096, 256, 0, stream>>>(xpm, ffo, (float*)d_out);
}
